// MaskTransformer_44109314130724
// MI455X (gfx1250) — matrix operand order repaired and checked by reading, not compiled, not run on hardware
//
#include <hip/hip_runtime.h>
#include <hip/hip_bf16.h>

// ---- problem constants ------------------------------------------------------
#define B_     32
#define L_     85
#define D_     768
#define H_     8
#define DK_    96
#define F_     3072
#define DEPTH_ 8
#define M_     (B_ * L_)          // 2720 rows, divisible by 32

typedef __attribute__((ext_vector_type(16))) __bf16 v16bf;
typedef __attribute__((ext_vector_type(8)))  float  v8f;
typedef unsigned int u32x4 __attribute__((ext_vector_type(4)));
typedef int          i32x4 __attribute__((ext_vector_type(4)));
typedef int          i32x8 __attribute__((ext_vector_type(8)));

union Frag { v16bf v; unsigned int u[8]; };

__device__ __forceinline__ unsigned short f2bf(float f) {
    unsigned int u = __float_as_uint(f);
    u += 0x7FFFu + ((u >> 16) & 1u);          // round-to-nearest-even
    return (unsigned short)(u >> 16);
}

// A-fragment K base for 16-bit A 16x32 (contiguous order k = 16h + 2i): dword i, lane-half h
__device__ __forceinline__ int amap(int i, int half) {
    return (half << 4) + (i << 1);
}
__device__ __forceinline__ unsigned int lds32(const unsigned short* p) {
    return *(const unsigned int*)p;
}

// ---- Tensor Data Mover: async 2D bf16 tile (global -> LDS), zero-fill OOB ---
#if defined(__has_builtin)
#if __has_builtin(__builtin_amdgcn_tensor_load_to_lds) && \
    __has_builtin(__builtin_amdgcn_s_wait_tensorcnt)
#define HAVE_TDM 1
#endif
#endif

#ifdef HAVE_TDM
__device__ __forceinline__ unsigned lds_off_of(const void* p) {
    // flat LDS address: low 32 bits are the LDS byte offset (ISA 10.2)
    return (unsigned)(unsigned long long)p;
}
__device__ __forceinline__ void tdm_load_2d(
        unsigned lds_off, const void* gaddr,
        unsigned tile_d0, unsigned tile_d1,     // tile (elements of 2B)
        unsigned tensor_d0, unsigned tensor_d1, // bounds for zero-fill
        unsigned stride0)                       // row stride (elements)
{
    unsigned long long ga = (unsigned long long)gaddr;
    u32x4 g0;
    g0.x = 1u;                                            // count=1, user D#
    g0.y = lds_off;                                       // lds_addr
    g0.z = (unsigned)ga;                                  // global_addr lo
    g0.w = (unsigned)((ga >> 32) & 0x01FFFFFFull) | (2u << 30); // addr hi | type=2
    i32x8 g1;
    g1[0] = (int)(1u << 16);                              // data_size=1 (2 bytes)
    g1[1] = (int)(tensor_d0 << 16);                       // tensor_dim0[15:0]
    g1[2] = (int)((tensor_d0 >> 16) | (tensor_d1 << 16)); // dim0 hi | dim1 lo
    g1[3] = (int)((tensor_d1 >> 16) | (tile_d0 << 16));   // dim1 hi | tile_dim0
    g1[4] = (int)(tile_d1 & 0xFFFFu);                     // tile_dim1 (tile_dim2=0)
    g1[5] = (int)stride0;                                 // tensor_dim0_stride lo32
    g1[6] = 0;
    g1[7] = 0;
    i32x4 z4 = {0, 0, 0, 0};
#if defined(__clang_major__) && (__clang_major__ >= 23)
    i32x8 z8 = {0, 0, 0, 0, 0, 0, 0, 0};
    __builtin_amdgcn_tensor_load_to_lds(g0, g1, z4, z4, z8, 0);
#else
    __builtin_amdgcn_tensor_load_to_lds(g0, g1, z4, z4, 0);
#endif
}
#endif // HAVE_TDM

// ---- InstanceNorm over last dim (D=768), fp32 -> bf16 -----------------------
__global__ __launch_bounds__(256) void inorm_kernel(const float* __restrict__ xc,
                                                    unsigned short* __restrict__ x2) {
    int row = blockIdx.x;
    int tid = threadIdx.x;
    const float* p = xc + (size_t)row * D_;
    float v0 = p[tid], v1 = p[tid + 256], v2 = p[tid + 512];
    __shared__ float rs[256], rss[256];
    rs[tid]  = v0 + v1 + v2;
    rss[tid] = v0 * v0 + v1 * v1 + v2 * v2;
    __syncthreads();
    for (int off = 128; off > 0; off >>= 1) {
        if (tid < off) { rs[tid] += rs[tid + off]; rss[tid] += rss[tid + off]; }
        __syncthreads();
    }
    float mean = rs[0] * (1.0f / D_);
    float var  = rss[0] * (1.0f / D_) - mean * mean;
    float inv  = rsqrtf(var + 1e-5f);
    unsigned short* o = x2 + (size_t)row * D_;
    o[tid]       = f2bf((v0 - mean) * inv);
    o[tid + 256] = f2bf((v1 - mean) * inv);
    o[tid + 512] = f2bf((v2 - mean) * inv);
}

// ---- weight convert + transpose: fp32 [K,N] -> bf16 [N,K] -------------------
__global__ __launch_bounds__(256) void wcvt_kernel(const float* __restrict__ W,
                                                   unsigned short* __restrict__ Wt,
                                                   int N, int K) {
    __shared__ float t[32][33];
    int k0 = blockIdx.x * 32, n0 = blockIdx.y * 32;
    int tx = threadIdx.x & 31, ty = threadIdx.x >> 5;   // 32 x 8
#pragma unroll
    for (int i = 0; i < 32; i += 8)
        t[ty + i][tx] = W[(size_t)(k0 + ty + i) * N + n0 + tx];
    __syncthreads();
#pragma unroll
    for (int i = 0; i < 32; i += 8)
        Wt[(size_t)(n0 + ty + i) * K + k0 + tx] = f2bf(t[tx][ty + i]);
}

// ---- generic WMMA GEMM: C[M,N] = A[M,K](bf16) @ Wt[N,K](bf16)^T + bias ------
// Block: 256 threads (8 waves) -> 32x128 tile, K-step 64 (4 WMMA / sync round).
// TDM path: double-buffered tensor_load_to_lds issued by wave 0.
#define MODE_BF16 0   // out = bf16(C)
#define MODE_GELU 1   // out = bf16(gelu(C))
#define MODE_ACC  2   // out(f32) += C   (residual accumulate)

__global__ __launch_bounds__(256) void gemm_kernel(
        const unsigned short* __restrict__ A,    // bf16 [M,K] row-major
        const unsigned short* __restrict__ Wt,   // bf16 [N,K] row-major (pre-transposed)
        const float*          __restrict__ bias, // f32  [N]
        void* __restrict__ out, int N, int K, int mode) {
#ifdef HAVE_TDM
    __shared__ __align__(16) unsigned short As[2][32 * 64];
    __shared__ __align__(16) unsigned short Ws[2][128 * 64];
#else
    __shared__ __align__(16) unsigned short As1[32 * 64];
    __shared__ __align__(16) unsigned short Ws1[128 * 64];
#endif

    int tid  = threadIdx.x;
    int lane = tid & 31, wid = tid >> 5;
    int wr   = wid >> 2, wc = wid & 3;         // 2x4 wave grid, 16x32 per wave
    int half = lane >> 4, col = lane & 15;
    int m0   = blockIdx.y * 32;
    int nb0  = blockIdx.x * 128;

    v8f acc0 = {0.f, 0.f, 0.f, 0.f, 0.f, 0.f, 0.f, 0.f};
    v8f acc1 = {0.f, 0.f, 0.f, 0.f, 0.f, 0.f, 0.f, 0.f};

#ifdef HAVE_TDM
    if (wid == 0) {
        tdm_load_2d(lds_off_of(As[0]), A  + (size_t)m0  * K, 64, 32,  64, 32,  (unsigned)K);
        tdm_load_2d(lds_off_of(Ws[0]), Wt + (size_t)nb0 * K, 64, 128, 64, 128, (unsigned)K);
    }
    int cur = 0;
#else
    int sa_r = tid >> 3, sa_c = (tid & 7) * 8;     // A: 8 elems/thread
    int sw_n = tid >> 1, sw_c = (tid & 1) * 32;    // W: 32 elems/thread
#endif

    for (int k0 = 0; k0 < K; k0 += 64) {
#ifdef HAVE_TDM
        if (wid == 0) {
            if (k0 + 64 < K) {
                tdm_load_2d(lds_off_of(As[cur ^ 1]), A  + (size_t)m0  * K + k0 + 64,
                            64, 32, 64, 32, (unsigned)K);
                tdm_load_2d(lds_off_of(Ws[cur ^ 1]), Wt + (size_t)nb0 * K + k0 + 64,
                            64, 128, 64, 128, (unsigned)K);
                __builtin_amdgcn_s_wait_tensorcnt(2);  // in-order: current pair done
            } else {
                __builtin_amdgcn_s_wait_tensorcnt(0);
            }
        }
        __syncthreads();
        const unsigned short* Ap = As[cur];
        const unsigned short* Wp = Ws[cur];
#else
        *(uint4*)&As1[sa_r * 64 + sa_c] =
            *(const uint4*)(A + (size_t)(m0 + sa_r) * K + k0 + sa_c);
        {
            const unsigned short* src = Wt + (size_t)(nb0 + sw_n) * K + k0 + sw_c;
            *(uint4*)&Ws1[sw_n * 64 + sw_c]      = *(const uint4*)(src);
            *(uint4*)&Ws1[sw_n * 64 + sw_c + 8]  = *(const uint4*)(src + 8);
            *(uint4*)&Ws1[sw_n * 64 + sw_c + 16] = *(const uint4*)(src + 16);
            *(uint4*)&Ws1[sw_n * 64 + sw_c + 24] = *(const uint4*)(src + 24);
        }
        if (k0 + 64 < K)
            __builtin_prefetch(Wt + (size_t)(nb0 + sw_n) * K + k0 + 64, 0, 1);
        __syncthreads();
        const unsigned short* Ap = As1;
        const unsigned short* Wp = Ws1;
#endif

#pragma unroll
        for (int ckk = 0; ckk < 2; ckk++) {
            int ck = ckk * 32;
            Frag a, b0, b1;
#pragma unroll
            for (int i = 0; i < 8; i++)
                a.u[i] = lds32(&Ap[(wr * 16 + col) * 64 + ck + amap(i, half)]);
#pragma unroll
            for (int i = 0; i < 8; i++)
                b0.u[i] = lds32(&Wp[(wc * 32 + col) * 64 + ck + (half << 4) + (i << 1)]);
#pragma unroll
            for (int i = 0; i < 8; i++)
                b1.u[i] = lds32(&Wp[(wc * 32 + 16 + col) * 64 + ck + (half << 4) + (i << 1)]);
            acc0 = __builtin_amdgcn_wmma_f32_16x16x32_bf16(
                    false, a.v, false, b0.v, (short)0, acc0, false, false);
            acc1 = __builtin_amdgcn_wmma_f32_16x16x32_bf16(
                    false, a.v, false, b1.v, (short)0, acc1, false, false);
        }
        __syncthreads();
#ifdef HAVE_TDM
        cur ^= 1;
#endif
    }

    // epilogue — C layout: lane = N col (mod 16), VGPR r = M row (+8 hi half)
    int mbase = m0 + wr * 16 + half * 8;
#pragma unroll
    for (int t = 0; t < 2; t++) {
        v8f acc = t ? acc1 : acc0;
        int n   = nb0 + wc * 32 + t * 16 + col;
        float bs = bias ? bias[n] : 0.f;
        if (mode == MODE_ACC) {
            float* of = (float*)out;
#pragma unroll
            for (int r = 0; r < 8; r++) {
                size_t idx = (size_t)(mbase + r) * N + n;
                of[idx] += acc[r] + bs;
            }
        } else if (mode == MODE_GELU) {
            unsigned short* ob = (unsigned short*)out;
#pragma unroll
            for (int r = 0; r < 8; r++) {
                float v = acc[r] + bs;
                v = 0.5f * v * (1.f + erff(v * 0.70710678118f));
                ob[(size_t)(mbase + r) * N + n] = f2bf(v);
            }
        } else {
            unsigned short* ob = (unsigned short*)out;
#pragma unroll
            for (int r = 0; r < 8; r++)
                ob[(size_t)(mbase + r) * N + n] = f2bf(acc[r] + bs);
        }
    }
}

// ---- masked attention for one (b, h, 16-query tile) -------------------------
__global__ __launch_bounds__(256) void attn_kernel(
        const unsigned short* __restrict__ qb,   // bf16 [B,L,D]
        const unsigned short* __restrict__ kb,
        const unsigned short* __restrict__ vb,
        const int*            __restrict__ mask, // int32 [B,L,L]
        unsigned short*       __restrict__ ao,   // bf16 [B,L,D] attn output
        float*                __restrict__ sout) // f32 [B,H,L,L] slice
{
    __shared__ __align__(16) unsigned short Qs[16 * 96];   // [q][dk]
    __shared__ __align__(16) unsigned short Ks[96 * 96];   // [key][dk]
    __shared__ __align__(16) unsigned short Vt[96 * 96];   // [dk][key]
    __shared__ __align__(16) unsigned short Pb[16 * 96];   // [q][key] bf16 probs
    __shared__ float S[16 * 96];
    __shared__ float red[16 * 16];
    __shared__ float rowsum_s[16], rowmax_s[16];

    int tid  = threadIdx.x;
    int lane = tid & 31, wid = tid >> 5;
    int half = lane >> 4, col = lane & 15;
    int qt   = blockIdx.x % 6;
    int bh   = blockIdx.x / 6;
    int h    = bh % H_;
    int b    = bh / H_;
    int hoff = h * DK_;

#ifdef HAVE_TDM
    if (wid == 0) {
        int q0 = qt * 16;
        tdm_load_2d(lds_off_of(Qs), qb + (size_t)(b * L_ + q0) * D_ + hoff,
                    96, 16, 96, (unsigned)(L_ - q0), D_);
        tdm_load_2d(lds_off_of(Ks), kb + (size_t)(b * L_) * D_ + hoff,
                    96, 96, 96, L_, D_);
    }
#else
    for (int idx = tid; idx < 16 * 96; idx += 256) {
        int r = idx / 96, dk = idx % 96, q = qt * 16 + r;
        Qs[idx] = (q < L_) ? qb[(size_t)(b * L_ + q) * D_ + hoff + dk] : (unsigned short)0;
    }
    for (int idx = tid; idx < 96 * 96; idx += 256) {
        int key = idx / 96, dk = idx % 96;
        Ks[idx] = (key < L_) ? kb[(size_t)(b * L_ + key) * D_ + hoff + dk] : (unsigned short)0;
    }
#endif
    // V transposed (TDM cannot transpose): Vt[dk][key]
    for (int idx = tid; idx < 96 * 96; idx += 256) {
        int dk = idx / 96, key = idx % 96;
        Vt[idx] = (key < L_) ? vb[(size_t)(b * L_ + key) * D_ + hoff + dk] : (unsigned short)0;
    }
#ifdef HAVE_TDM
    if (wid == 0) __builtin_amdgcn_s_wait_tensorcnt(0);
#endif
    __syncthreads();

    // ---- scores S[16 q][96 key]: waves 0..5 each do one 16-key tile ----
    if (wid < 6) {
        v8f acc = {0.f, 0.f, 0.f, 0.f, 0.f, 0.f, 0.f, 0.f};
#pragma unroll
        for (int c = 0; c < 3; c++) {
            int kc = c * 32;
            Frag a, bf;
#pragma unroll
            for (int i = 0; i < 8; i++)
                a.u[i] = lds32(&Qs[col * 96 + kc + amap(i, half)]);
#pragma unroll
            for (int i = 0; i < 8; i++)
                bf.u[i] = lds32(&Ks[(wid * 16 + col) * 96 + kc + (half << 4) + (i << 1)]);
            acc = __builtin_amdgcn_wmma_f32_16x16x32_bf16(
                    false, a.v, false, bf.v, (short)0, acc, false, false);
        }
#pragma unroll
        for (int r = 0; r < 8; r++)
            S[(r + 8 * half) * 96 + wid * 16 + col] = acc[r];
    }
    __syncthreads();

    // ---- scale + mask + softmax over 85 keys (16 threads per query row) ----
    int r  = tid >> 4, c0 = tid & 15;
    int q  = qt * 16 + r;
    bool valid = (q < L_);
    const float scale = 0.1020620726f;        // 1/sqrt(96)
    const int* mrow = mask + (size_t)(b * L_ + (valid ? q : 0)) * L_;
    float vals[6];
    float mx = -3.4e38f;
#pragma unroll
    for (int j = 0; j < 6; j++) {
        int c = c0 + j * 16;
        float s = -3.4e38f;
        if (valid && c < L_) {
            s = S[r * 96 + c] * scale;
            if (mrow[c] == 1) s = -10000.f;
        }
        vals[j] = s;
        mx = fmaxf(mx, s);
    }
    red[r * 16 + c0] = mx;
    __syncthreads();
    if (c0 == 0) {
        float m = -3.4e38f;
        for (int i = 0; i < 16; i++) m = fmaxf(m, red[r * 16 + i]);
        rowmax_s[r] = m;
    }
    __syncthreads();
    float rm = rowmax_s[r], sum = 0.f;
#pragma unroll
    for (int j = 0; j < 6; j++) {
        float e = (vals[j] > -3.0e38f) ? __expf(vals[j] - rm) : 0.f;
        vals[j] = e;
        sum += e;
    }
    red[r * 16 + c0] = sum;
    __syncthreads();
    if (c0 == 0) {
        float t = 0.f;
        for (int i = 0; i < 16; i++) t += red[r * 16 + i];
        rowsum_s[r] = t;
    }
    __syncthreads();
    float inv = rowsum_s[r] > 0.f ? 1.f / rowsum_s[r] : 0.f;
    float* srow = sout + ((size_t)(b * H_ + h) * L_ + (valid ? q : 0)) * L_;
#pragma unroll
    for (int j = 0; j < 6; j++) {
        int c = c0 + j * 16;                  // covers 0..95
        float p = vals[j] * inv;
        bool live = valid && (c < L_);
        Pb[r * 96 + c] = live ? f2bf(p) : (unsigned short)0;
        if (live) srow[c] = p;
    }
    __syncthreads();

    // ---- O[16 q][96 dk] = P @ V : waves 0..5 each do one 16-dk tile ----
    if (wid < 6) {
        v8f acc = {0.f, 0.f, 0.f, 0.f, 0.f, 0.f, 0.f, 0.f};
#pragma unroll
        for (int c = 0; c < 3; c++) {
            int kc = c * 32;
            Frag a, bf;
#pragma unroll
            for (int i = 0; i < 8; i++)
                a.u[i] = lds32(&Pb[col * 96 + kc + amap(i, half)]);
#pragma unroll
            for (int i = 0; i < 8; i++)
                bf.u[i] = lds32(&Vt[(wid * 16 + col) * 96 + kc + (half << 4) + (i << 1)]);
            acc = __builtin_amdgcn_wmma_f32_16x16x32_bf16(
                    false, a.v, false, bf.v, (short)0, acc, false, false);
        }
#pragma unroll
        for (int rr = 0; rr < 8; rr++) {
            int qq = qt * 16 + rr + 8 * half;
            if (qq < L_)
                ao[(size_t)(b * L_ + qq) * D_ + hoff + wid * 16 + col] = f2bf(acc[rr]);
        }
    }
}

__global__ __launch_bounds__(256) void copy_f32(const float* __restrict__ in,
                                                float* __restrict__ out, int n) {
    int i = blockIdx.x * 256 + threadIdx.x;
    if (i < n) out[i] = in[i];
}

// ---- host orchestration -----------------------------------------------------
extern "C" void kernel_launch(void* const* d_in, const int* in_sizes, int n_in,
                              void* d_out, int out_size, void* d_ws, size_t ws_size,
                              hipStream_t stream) {
    (void)in_sizes; (void)n_in; (void)out_size; (void)ws_size;

    const float* x = (const float*)d_in[0];
    const float* Wb[4][4]; const float* bb[4][4];
    for (int br = 0; br < 4; br++)
        for (int p = 0; p < 4; p++) {
            Wb[br][p] = (const float*)d_in[1 + br * 8 + p * 2];
            bb[br][p] = (const float*)d_in[1 + br * 8 + p * 2 + 1];
        }
    const float* ffW1 = (const float*)d_in[33];
    const float* ffb1 = (const float*)d_in[34];
    const float* ffW2 = (const float*)d_in[35];
    const float* ffb2 = (const float*)d_in[36];
    const int* masks[4] = {(const int*)d_in[37], (const int*)d_in[38],
                           (const int*)d_in[39], (const int*)d_in[40]};

    char* ws = (char*)d_ws;
    size_t off = 0;
    auto alloc = [&](size_t bytes) {
        void* p = ws + off;
        off += (bytes + 255) & ~(size_t)255;
        return p;
    };
    float*          xc    = (float*)         alloc((size_t)M_ * D_ * 4);
    unsigned short* x2    = (unsigned short*)alloc((size_t)M_ * D_ * 2);
    unsigned short* qbuf  = (unsigned short*)alloc((size_t)M_ * D_ * 2);
    unsigned short* kbuf  = (unsigned short*)alloc((size_t)M_ * D_ * 2);
    unsigned short* vbuf  = (unsigned short*)alloc((size_t)M_ * D_ * 2);
    unsigned short* aobf  = (unsigned short*)alloc((size_t)M_ * D_ * 2);
    unsigned short* hbuf  = (unsigned short*)alloc((size_t)M_ * F_ * 2);
    unsigned short* wtbuf = (unsigned short*)alloc((size_t)D_ * F_ * 2); // largest W

    float* xout = (float*)d_out;
    float* sbase = xout + (size_t)M_ * D_;
    const size_t SLayer  = (size_t)B_ * H_ * L_ * L_;
    const size_t SBranch = (size_t)DEPTH_ * SLayer;

    const int nelem = M_ * D_;
    copy_f32<<<(nelem + 255) / 256, 256, 0, stream>>>(x, xc, nelem);

    dim3 gN768(D_ / 128, M_ / 32), gN3072(F_ / 128, M_ / 32);
    dim3 cvtDD(D_ / 32, D_ / 32);      // 768x768
    dim3 cvtDF(D_ / 32, F_ / 32);      // W1: K=768, N=3072
    dim3 cvtFD(F_ / 32, D_ / 32);      // W2: K=3072, N=768

    for (int layer = 0; layer < DEPTH_; layer++) {
        inorm_kernel<<<M_, 256, 0, stream>>>(xc, x2);
        for (int br = 0; br < 4; br++) {
            const float* Wq = Wb[br][0] + (size_t)layer * D_ * D_;
            const float* Wk = Wb[br][1] + (size_t)layer * D_ * D_;
            const float* Wv = Wb[br][2] + (size_t)layer * D_ * D_;
            const float* Wo = Wb[br][3] + (size_t)layer * D_ * D_;
            const float* bq = bb[br][0] + (size_t)layer * D_;
            const float* bk = bb[br][1] + (size_t)layer * D_;
            const float* bv = bb[br][2] + (size_t)layer * D_;
            const float* bo = bb[br][3] + (size_t)layer * D_;
            wcvt_kernel<<<cvtDD, 256, 0, stream>>>(Wq, wtbuf, D_, D_);
            gemm_kernel<<<gN768, 256, 0, stream>>>(x2, wtbuf, bq, qbuf, D_, D_, MODE_BF16);
            wcvt_kernel<<<cvtDD, 256, 0, stream>>>(Wk, wtbuf, D_, D_);
            gemm_kernel<<<gN768, 256, 0, stream>>>(x2, wtbuf, bk, kbuf, D_, D_, MODE_BF16);
            wcvt_kernel<<<cvtDD, 256, 0, stream>>>(Wv, wtbuf, D_, D_);
            gemm_kernel<<<gN768, 256, 0, stream>>>(x2, wtbuf, bv, vbuf, D_, D_, MODE_BF16);
            attn_kernel<<<B_ * H_ * 6, 256, 0, stream>>>(
                qbuf, kbuf, vbuf, masks[br], aobf,
                sbase + (size_t)br * SBranch + (size_t)layer * SLayer);
            wcvt_kernel<<<cvtDD, 256, 0, stream>>>(Wo, wtbuf, D_, D_);
            gemm_kernel<<<gN768, 256, 0, stream>>>(aobf, wtbuf, bo, xc, D_, D_, MODE_ACC);
        }
        inorm_kernel<<<M_, 256, 0, stream>>>(xc, x2);
        wcvt_kernel<<<cvtDF, 256, 0, stream>>>(ffW1 + (size_t)layer * D_ * F_, wtbuf, F_, D_);
        gemm_kernel<<<gN3072, 256, 0, stream>>>(x2, wtbuf, ffb1 + (size_t)layer * F_,
                                                hbuf, F_, D_, MODE_GELU);
        wcvt_kernel<<<cvtFD, 256, 0, stream>>>(ffW2 + (size_t)layer * F_ * D_, wtbuf, D_, F_);
        gemm_kernel<<<gN768, 256, 0, stream>>>(hbuf, wtbuf, ffb2 + (size_t)layer * D_,
                                               xc, D_, F_, MODE_ACC);
    }
    copy_f32<<<(nelem + 255) / 256, 256, 0, stream>>>(xc, xout, nelem);
}
